// GraphAttentionLayer_87668872446709
// MI455X (gfx1250) — matrix operand order repaired and checked by reading, not compiled, not run on hardware
//
#include <hip/hip_runtime.h>

typedef __attribute__((ext_vector_type(16))) __bf16 v16bf;
typedef __attribute__((ext_vector_type(8)))  __bf16 v8bf;
typedef __attribute__((ext_vector_type(8)))  float  v8f;
typedef __attribute__((ext_vector_type(4)))  float  v4f;

#define GN    6144
#define IN_F  512
#define OUT_F 512
#define NHEAD 8
#define DHEAD 64
#define ALPHA 0.2f
#define NEGV  (-1e9f)
#define LOG2E 1.44269504088896340736f

// ---------------------------------------------------------------------------
// Kernel 0: bit-pack adjacency. adjbits[row][w] bit k = (adj[row][w*32+k] > 0)
// One wave per row-slice via wave ballot; fully coalesced reads.
// ---------------------------------------------------------------------------
__global__ __launch_bounds__(256) void gat_pack(const int* __restrict__ adj,
                                                unsigned* __restrict__ bits) {
  const int wave = threadIdx.x >> 5;
  const int lane = threadIdx.x & 31;
  const int row = blockIdx.x * 8 + wave;
  const int* ar = adj + (size_t)row * GN;
  unsigned* br = bits + (size_t)row * (GN / 32);
  for (int w = 0; w < GN / 32; ++w) {
    int v = ar[w * 32 + lane];
    unsigned word = __builtin_amdgcn_ballot_w32(v > 0);
    if (lane == 0) br[w] = word;
  }
}

// ---------------------------------------------------------------------------
// Kernel 1: hbT[o][i] = bf16( (x @ W.T + b)[i][o] )   (d-major / transposed)
// One wave per 16x16 tile, v_wmma_f32_16x16x32_bf16, K-loop of 32.
// ---------------------------------------------------------------------------
__global__ __launch_bounds__(256) void gat_gemm_hbT(
    const float* __restrict__ x, const float* __restrict__ W,
    const float* __restrict__ b, __bf16* __restrict__ hbT) {
  const int wave = threadIdx.x >> 5;
  const int lane = threadIdx.x & 31;
  const int tile = blockIdx.x * 8 + wave;            // 12288 tiles total
  const int ti = tile % (GN / 16);
  const int to = tile / (GN / 16);
  const int i0 = ti * 16, o0 = to * 16;
  const int rc    = lane & 15;                       // A-row / B-col / D-col
  const int hi    = (lane >> 4) & 1;
  const int koffA = hi * 8;
  const int koffB = hi * 16;

  v8f acc = {};
  for (int k0 = 0; k0 < IN_F; k0 += 32) {
    const float* xr = x + (size_t)(i0 + rc) * IN_F + k0;
    v4f a0 = *(const v4f*)(xr + koffA);
    v4f a1 = *(const v4f*)(xr + koffA + 4);
    v4f a2 = *(const v4f*)(xr + 16 + koffA);
    v4f a3 = *(const v4f*)(xr + 16 + koffA + 4);
    v16bf A;
#pragma unroll
    for (int e = 0; e < 4; ++e) {
      A[e]      = (__bf16)a0[e];
      A[4 + e]  = (__bf16)a1[e];
      A[8 + e]  = (__bf16)a2[e];
      A[12 + e] = (__bf16)a3[e];
    }
    const float* wr = W + (size_t)(o0 + rc) * IN_F + k0;
    v4f b0 = *(const v4f*)(wr + koffA);
    v4f b1 = *(const v4f*)(wr + koffA + 4);
    v4f b2 = *(const v4f*)(wr + 16 + koffA);
    v4f b3 = *(const v4f*)(wr + 16 + koffA + 4);
    v16bf B;
#pragma unroll
    for (int e = 0; e < 4; ++e) {
      B[e]      = (__bf16)b0[e];
      B[4 + e]  = (__bf16)b1[e];
      B[8 + e]  = (__bf16)b2[e];
      B[12 + e] = (__bf16)b3[e];
    }
    acc = __builtin_amdgcn_wmma_f32_16x16x32_bf16(
        false, A, false, B, (short)0, acc, false, false);
  }

  const float bias = b[o0 + rc];
  v8bf op;
#pragma unroll
  for (int r = 0; r < 8; ++r) op[r] = (__bf16)(acc[r] + bias);
  *(v8bf*)(hbT + (size_t)(o0 + rc) * GN + i0 + hi * 8) = op;
}

// ---------------------------------------------------------------------------
// Kernel 2: attention coefficients, pre-scaled by log2(e) so K3 can use a
// bare v_exp_f32 (2^x). leaky/max commute with the positive scale.
// ---------------------------------------------------------------------------
__global__ __launch_bounds__(256) void gat_coef(
    const __bf16* __restrict__ hbT, const float* __restrict__ aw,
    float* __restrict__ a0, float* __restrict__ a1) {
  const int n = blockIdx.x * blockDim.x + threadIdx.x;
  const int h = blockIdx.y;
  float s0 = 0.f, s1 = 0.f;
#pragma unroll 4
  for (int d = 0; d < DHEAD; ++d) {
    float v = (float)hbT[(size_t)(h * DHEAD + d) * GN + n];
    s0 = fmaf(v, aw[(h * DHEAD + d) * 2 + 0], s0);
    s1 = fmaf(v, aw[(h * DHEAD + d) * 2 + 1], s1);
  }
  a0[h * GN + n] = s0 * LOG2E;
  a1[h * GN + n] = s1 * LOG2E;
}

// ---------------------------------------------------------------------------
// Kernel 3: fused masked-softmax attention aggregation (flash style, base-2).
// Block = 8 waves = 8 heads sharing one 16-row i-block. Per 32-wide j chunk:
// 1 b32 mask load + 4 b128 a1 loads, masked leaky scores in WMMA A-layout,
// online softmax with ballot-gated (rare) accumulator rescale, then
// 4x v_wmma_f32_16x16x32_bf16 against d-major hbT B-tiles.
// ---------------------------------------------------------------------------
__global__ __launch_bounds__(256) void gat_attn(
    const unsigned* __restrict__ adjb, const __bf16* __restrict__ hbT,
    const float* __restrict__ a0v, const float* __restrict__ a1v,
    float* __restrict__ out) {
  const int head = threadIdx.x >> 5;
  const int lane = threadIdx.x & 31;
  const int i0 = blockIdx.x * 16;
  const int row   = lane & 15;
  const int hi    = (lane >> 4) & 1;
  const int hioff = hi * 8;
  const int koffA = hi * 8;
  const int koffB = hi * 16;
  const int hd0 = head * DHEAD;

  const float a0i = a0v[head * GN + i0 + row];
  const float* a1p = a1v + head * GN;
  const unsigned* abr = adjb + (size_t)(i0 + row) * (GN / 32);

  float m_run = -INFINITY;
  float l_run = 0.f;
  v8f acc[4] = {};

  for (int j0 = 0; j0 < GN; j0 += 32) {
    // mask bits for this lane's 16 (row, j) slots live in one dword
    const unsigned bits = abr[j0 >> 5] >> koffA;   // element e -> bit e (+8 if e>=8)
    v4f c0 = *(const v4f*)(a1p + j0 + koffA);
    v4f c1 = *(const v4f*)(a1p + j0 + koffA + 4);
    v4f c2 = *(const v4f*)(a1p + j0 + 16 + koffA);
    v4f c3 = *(const v4f*)(a1p + j0 + 16 + koffA + 4);

    float s[16];
#pragma unroll
    for (int e = 0; e < 4; ++e) {
      s[e] = c0[e]; s[4 + e] = c1[e]; s[8 + e] = c2[e]; s[12 + e] = c3[e];
    }
    float mt = -INFINITY;
#pragma unroll
    for (int e = 0; e < 16; ++e) {
      float t = a0i + s[e];
      t = fmaxf(t, ALPHA * t);                       // leaky_relu, 2 ops
      const unsigned bit = 1u << (e + ((e < 8) ? 0 : 8));
      s[e] = (bits & bit) ? t : NEGV;
      mt = fmaxf(mt, s[e]);
    }
    // combine row-max across the two half-wave copies of each row
    mt = fmaxf(mt, __shfl_xor(mt, 16, 32));
    const float m_new = fmaxf(m_run, mt);

    // rare path: only rescale when some row's max actually increased
    if (__builtin_amdgcn_ballot_w32(m_new > m_run) != 0u) {
      const float scale = __builtin_exp2f(m_run - m_new);  // 0 on first iter
      l_run *= scale;
#pragma unroll
      for (int r = 0; r < 8; ++r) {
        float sr = __shfl(scale, r + hioff, 32);
#pragma unroll
        for (int t = 0; t < 4; ++t) acc[t][r] *= sr;
      }
    }
    m_run = m_new;

    // p = 2^(s - m): masked slots underflow to exactly 0 (s = -1e9)
    float p[16];
    float psum = 0.f;
#pragma unroll
    for (int e = 0; e < 16; ++e) {
      p[e] = __builtin_exp2f(s[e] - m_new);
      psum += p[e];
    }
    l_run += psum;

    v16bf P;
#pragma unroll
    for (int e = 0; e < 16; ++e) P[e] = (__bf16)p[e];

#pragma unroll
    for (int t = 0; t < 4; ++t) {
      const __bf16* hp =
          hbT + (size_t)(hd0 + t * 16 + row) * GN + j0 + koffA;
      v8bf blo = *(const v8bf*)hp;
      v8bf bhi = *(const v8bf*)(hp + 16);
      v16bf B;
#pragma unroll
      for (int e = 0; e < 8; ++e) { B[e] = blo[e]; B[8 + e] = bhi[e]; }
      acc[t] = __builtin_amdgcn_wmma_f32_16x16x32_bf16(
          false, P, false, B, (short)0, acc[t], false, false);
    }
  }

  // Finalize. Guard: if a row never saw a real neighbor its max is still
  // ~NEGV -> reference post-mask makes that row all-zero.
  float l_comb = l_run + __shfl_xor(l_run, 16, 32);
  if (m_run < -0.25e9f) l_comb = 0.f;
#pragma unroll
  for (int r = 0; r < 8; ++r) {
    float lr = __shfl(l_comb, r + hioff, 32);
    float inv = (lr > 0.f) ? (1.f / lr) : 0.f;
    const int i = i0 + r + hioff;
#pragma unroll
    for (int t = 0; t < 4; ++t) {
      out[(size_t)i * OUT_F + hd0 + t * 16 + row] = acc[t][r] * inv;
    }
  }
}

// ---------------------------------------------------------------------------
extern "C" void kernel_launch(void* const* d_in, const int* in_sizes, int n_in,
                              void* d_out, int out_size, void* d_ws,
                              size_t ws_size, hipStream_t stream) {
  const float* x   = (const float*)d_in[0];
  const int*   adj = (const int*)d_in[1];
  const float* W   = (const float*)d_in[2];
  const float* b   = (const float*)d_in[3];
  const float* aw  = (const float*)d_in[4];
  float* out = (float*)d_out;

  char* ws = (char*)d_ws;
  __bf16* hbT = (__bf16*)ws;                                   // 6,291,456 B
  size_t off = (size_t)OUT_F * GN * sizeof(__bf16);
  float* a0 = (float*)(ws + off);  off += (size_t)NHEAD * GN * sizeof(float);
  float* a1 = (float*)(ws + off);  off += (size_t)NHEAD * GN * sizeof(float);
  unsigned* adjb = (unsigned*)(ws + off);                      // 4,718,592 B

  gat_pack<<<GN / 8, 256, 0, stream>>>(adj, adjb);
  gat_gemm_hbT<<<(GN / 16) * (OUT_F / 16) / 8, 256, 0, stream>>>(x, W, b, hbT);
  gat_coef<<<dim3(GN / 256, NHEAD), 256, 0, stream>>>(hbT, aw, a0, a1);
  gat_attn<<<GN / 16, 256, 0, stream>>>(adjb, hbT, a0, a1, out);
}
